// ShallowRegressionLSTM_75694503624729
// MI455X (gfx1250) — matrix operand order repaired and checked by reading, not compiled, not run on hardware
//
#include <hip/hip_runtime.h>

typedef __attribute__((ext_vector_type(16))) _Float16 v16h;
typedef __attribute__((ext_vector_type(8)))  float    v8f;
typedef __attribute__((ext_vector_type(8)))  unsigned v8u;

#define T_LEN 512
#define I_DIM 4
#define H_DIM 16

__device__ __forceinline__ float fast_tanh(float x) {
#if __has_builtin(__builtin_amdgcn_tanhf)
    return __builtin_amdgcn_tanhf(x);            // single v_tanh_f32 (CDNA5 TRANS op)
#elif __has_builtin(__builtin_amdgcn_tanh_f32)
    return __builtin_amdgcn_tanh_f32(x);
#else
    return 1.0f - 2.0f * __builtin_amdgcn_rcpf(1.0f + __builtin_exp2f(2.88539008178f * x));
#endif
}
// input is ALREADY 0.5*x (folded into the WMMA A-tiles)
__device__ __forceinline__ float sigmoid_prescaled(float halfx) {
    return fmaf(0.5f, fast_tanh(halfx), 0.5f);
}
__device__ __forceinline__ float lane_readf(float v, int srcLane) {
    return __int_as_float(__builtin_amdgcn_ds_bpermute(srcLane << 2, __float_as_int(v)));
}
__device__ __forceinline__ unsigned lane_readu(unsigned v, int srcLane) {
    return (unsigned)__builtin_amdgcn_ds_bpermute(srcLane << 2, (int)v);
}
// two f32 -> packed f16 pair in one v_cvt_pk_rtz_f16_f32
__device__ __forceinline__ unsigned pk16(float a, float b) {
    auto t = __builtin_amdgcn_cvt_pkrtz(a, b);   // __fp16 ext_vector_type(2)
    return __builtin_bit_cast(unsigned, t);
}

__global__ __launch_bounds__(256)
void lstm16_wmma_kernel(const float* __restrict__ x,
                        const float* __restrict__ W_ih,
                        const float* __restrict__ W_hh,
                        const float* __restrict__ b_ih,
                        const float* __restrict__ b_hh,
                        const float* __restrict__ W_out,
                        const float* __restrict__ b_out,
                        float* __restrict__ out,
                        int Btot)
{
    const int lane = threadIdx.x & 31;
    const int wave = threadIdx.x >> 5;
    const int m    = lane & 15;   // batch column within the 16-wide tile (also gate row for A)
    const int hi   = lane >> 4;   // which lane-half we are in
    const int tile = blockIdx.x * (blockDim.x >> 5) + wave;
    const int b0   = tile * 16;
    if (b0 >= Btot) return;

    // ------------------------------------------------------------------
    // 4 persistent A tiles (f16 16x32), one per gate group i,f,g,o.
    // 16-bit A layout: lanes 0-15 : elems 0..7 -> K=0..7, elems 8..15 -> K=16..23
    //                  lanes 16-31: elems 0..7 -> K=8..15, elems 8..15 -> K=24..31
    // K semantic: 0..15 = W_hh[:,k], 16..19 = W_ih[:,k], 20 = fused bias, rest 0.
    // i/f/o tiles (g = 0,1,3) are pre-scaled by 0.5 so sigmoid needs no input mul.
    // ------------------------------------------------------------------
    v16h A[4];
#pragma unroll
    for (int g = 0; g < 4; ++g) {
        const int   gate  = g * 16 + m;
        const float scale = (g == 2) ? 1.0f : 0.5f;   // tanh gate keeps full scale
        v16h a;
#pragma unroll
        for (int e = 0; e < 16; ++e) a[e] = (_Float16)0.0f;
        if (hi == 0) {
#pragma unroll
            for (int e = 0; e < 8; ++e) a[e] = (_Float16)(scale * W_hh[gate * H_DIM + e]);     // K=0..7
#pragma unroll
            for (int e = 0; e < 4; ++e) a[8 + e] = (_Float16)(scale * W_ih[gate * I_DIM + e]); // K=16..19
            a[12] = (_Float16)(scale * (b_ih[gate] + b_hh[gate]));                             // K=20 bias
        } else {
#pragma unroll
            for (int e = 0; e < 8; ++e) a[e] = (_Float16)(scale * W_hh[gate * H_DIM + 8 + e]); // K=8..15
        }
        A[g] = a;
    }

    // Output head weights: lane holds W_out for hidden rows r + 8*hi
    float wo[8];
#pragma unroll
    for (int r = 0; r < 8; ++r) wo[r] = W_out[r + 8 * hi];

    // LSTM state: lane (m,hi) holds h/c for batch (b0+m), hidden rows r + 8*hi
    float h[8], c[8];
#pragma unroll
    for (int r = 0; r < 8; ++r) { h[r] = 0.0f; c[r] = 0.0f; }

    const float4* __restrict__ xrow = (const float4*)x;  // x[b][t] is a float4
    const size_t rowbase = (size_t)(b0 + m) * T_LEN;

    float4 xv = xrow[rowbase];          // software pipeline: x_t for t=0
    const v8f zacc = {};                // C operand (bias already fused via K=20)

    for (int t = 0; t < T_LEN; ++t) {
        // prefetch one cacheline (8 timesteps) ahead, clamped in-bounds
        const int tp = (t + 8 < T_LEN) ? t + 8 : T_LEN - 1;
        __builtin_prefetch(&xrow[rowbase + tp], 0, 1);

        // issue next timestep's load early (independent of this iteration's math)
        const int tn = (t + 1 < T_LEN) ? t + 1 : t;
        float4 xn = xrow[rowbase + tn];

        // Pack this lane's h[0..8) into 4 f16-pair dwords, then pull the upper
        // hidden half (lane n+16's pack) across with 4 ds_bpermute ops.
        unsigned hp[4], hup[4];
#pragma unroll
        for (int d = 0; d < 4; ++d) hp[d] = pk16(h[2 * d], h[2 * d + 1]);
#pragma unroll
        for (int d = 0; d < 4; ++d) hup[d] = lane_readu(hp[d], m + 16);

        // ------------------------------------------------------------------
        // B matrix (f16 32x16): lanes 0-15 carry K=0..7 (h) and K=16..23 (x_t, 1.0 bias row, 0),
        // lanes 16-31 carry K=8..15 (h) and K=24..31 (zero padding).
        // Assemble dword-wise: 8 cndmask_b32 on the lane-half predicate.
        // ------------------------------------------------------------------
        v8u bl, bh;
        bl[0] = hp[0];  bl[1] = hp[1];  bl[2] = hp[2];  bl[3] = hp[3];
        bl[4] = pk16(xv.x, xv.y);       // K=16,17
        bl[5] = pk16(xv.z, xv.w);       // K=18,19
        bl[6] = 0x00003C00u;            // K=20 -> 1.0 (bias multiplier), K=21 -> 0
        bl[7] = 0u;
        bh[0] = hp[0];  bh[1] = hp[1];  bh[2] = hp[2];  bh[3] = hp[3];
        bh[4] = 0u; bh[5] = 0u; bh[6] = 0u; bh[7] = 0u;
        v8u bsel;
#pragma unroll
        for (int d = 0; d < 8; ++d) bsel[d] = hi ? bh[d] : bl[d];
        v16h Bm = __builtin_bit_cast(v16h, bsel);

        // All 64 gates for 16 batch rows: 4 WMMAs (h-proj + x-proj + bias fused)
        v8f gi = __builtin_amdgcn_wmma_f32_16x16x32_f16(false, A[0], false, Bm, (short)0, zacc, false, false);
        v8f gf = __builtin_amdgcn_wmma_f32_16x16x32_f16(false, A[1], false, Bm, (short)0, zacc, false, false);
        v8f gg = __builtin_amdgcn_wmma_f32_16x16x32_f16(false, A[2], false, Bm, (short)0, zacc, false, false);
        v8f go = __builtin_amdgcn_wmma_f32_16x16x32_f16(false, A[3], false, Bm, (short)0, zacc, false, false);

        // Elementwise LSTM cell update (co-executes with the XDL pipeline).
        // i/f/o accumulators are already 0.5*pre-activation.
#pragma unroll
        for (int r = 0; r < 8; ++r) {
            const float iv = sigmoid_prescaled(gi[r]);
            const float fv = sigmoid_prescaled(gf[r]);
            const float gv = fast_tanh(gg[r]);
            const float ov = sigmoid_prescaled(go[r]);
            c[r] = fmaf(fv, c[r], iv * gv);
            h[r] = ov * fast_tanh(c[r]);
        }

        xv = xn;
    }

    // Head: out[b] = h_T . W_out + b_out ; partial per lane-half, combine via bpermute
    float part = 0.0f;
#pragma unroll
    for (int r = 0; r < 8; ++r) part = fmaf(h[r], wo[r], part);
    const float tot = part + lane_readf(part, m + 16);
    if (lane < 16) out[b0 + m] = tot + b_out[0];
}

extern "C" void kernel_launch(void* const* d_in, const int* in_sizes, int n_in,
                              void* d_out, int out_size, void* d_ws, size_t ws_size,
                              hipStream_t stream) {
    const float* x     = (const float*)d_in[0];
    const float* W_ih  = (const float*)d_in[1];
    const float* W_hh  = (const float*)d_in[2];
    const float* b_ih  = (const float*)d_in[3];
    const float* b_hh  = (const float*)d_in[4];
    const float* W_out = (const float*)d_in[5];
    const float* b_out = (const float*)d_in[6];
    float* out = (float*)d_out;

    const int Btot = in_sizes[0] / (T_LEN * I_DIM);   // 16384
    const int wavesPerBlock = 8;                      // 256 threads
    const int tiles  = (Btot + 15) / 16;              // one wave per 16-batch tile
    const int blocks = (tiles + wavesPerBlock - 1) / wavesPerBlock;

    lstm16_wmma_kernel<<<blocks, wavesPerBlock * 32, 0, stream>>>(
        x, W_ih, W_hh, b_ih, b_hh, W_out, b_out, out, Btot);
}
